// MultiLoRALinear_25116968747193
// MI455X (gfx1250) — matrix operand order repaired and checked by reading, not compiled, not run on hardware
//
#include <hip/hip_runtime.h>

// ---------------------------------------------------------------------------
// MultiLoRA linear for MI455X (gfx1250, wave32, WMMA)
//   out[t,o] = x[t,:]·W[o,:] + b[o] + (alpha/R) * sum_r a[t,r]*lora_B[seg(t),o,r]
//   a[t,r]   = x[t,:]·lora_A[seg(t),r,:]
// Base GEMM: bf16 WMMA 16x16x32, f32 accumulate, 2-stage software pipeline
// with statically addressed LDS double buffers (branch-free inner loop),
// async-load-to-LDS for the LoRA activation tile.
// ---------------------------------------------------------------------------

typedef __attribute__((ext_vector_type(8)))  float  v8f;
typedef __attribute__((ext_vector_type(8)))  __bf16 v8bf;
typedef __attribute__((ext_vector_type(16))) __bf16 v16bf;

constexpr int T_TOK = 16384;
constexpr int IN_F  = 4096;
constexpr int OUT_F = 4096;
constexpr int RANK  = 16;
constexpr int NADP  = 8;
constexpr float SCALING = 16.0f / 16.0f;   // alpha / rank

constexpr int BM  = 128;   // token tile
constexpr int BN  = 128;   // out-feature tile
constexpr int BK  = 32;    // K step (one bf16 WMMA K)
constexpr int LDK = 40;    // padded LDS row stride in bf16 (80B, 16B aligned)

// ---------------------------------------------------------------------------
// Kernel 1: per-token rank-R A-projection, pre-scaled by alpha/R.
// One wave32 per token: lane = (r in 0..15, K-half in {0,1}).
// ---------------------------------------------------------------------------
__global__ __launch_bounds__(256) void lora_a_proj(
    const float* __restrict__ x, const float* __restrict__ lA,
    const int* __restrict__ offsets, float* __restrict__ a_ws)
{
    const int wave = (blockIdx.x * 256 + threadIdx.x) >> 5;   // token id
    const int lane = threadIdx.x & 31;
    const int t    = wave;

    int seg = 0;
    #pragma unroll
    for (int n = 0; n < NADP; ++n) seg += (t >= offsets[n]) ? 1 : 0;

    const int r = lane & 15;
    const int h = lane >> 4;

    const float4* xp = (const float4*)(x  + (size_t)t * IN_F + h * (IN_F / 2));
    const float4* ap = (const float4*)(lA + ((size_t)seg * RANK + r) * IN_F + h * (IN_F / 2));

    float acc = 0.0f;
    for (int it = 0; it < IN_F / 8; ++it) {
        float4 xv = xp[it];
        float4 av = ap[it];
        acc += xv.x * av.x + xv.y * av.y + xv.z * av.z + xv.w * av.w;
    }
    acc += __shfl_xor(acc, 16, 32);       // combine the two K-halves
    if (lane < 16) a_ws[(size_t)t * RANK + r] = SCALING * acc;
}

// ---------------------------------------------------------------------------
// Kernel 2: base GEMM on WMMA bf16 + fused bias + LoRA-B epilogue.
// Block = 256 threads = 8 waves (2x4 over a 128x128 tile); wave = 64x32.
// ---------------------------------------------------------------------------
__global__ __launch_bounds__(256) void lora_gemm(
    const float* __restrict__ x, const float* __restrict__ W,
    const float* __restrict__ bias, const float* __restrict__ lB,
    const int* __restrict__ offsets, const float* __restrict__ a_ws,
    float* __restrict__ out)
{
    __shared__ __align__(16) __bf16 sX[2][BM * LDK];  // token tiles, bf16
    __shared__ __align__(16) __bf16 sW[2][BN * LDK];  // weight tiles, bf16
    __shared__ __align__(16) float  sAc[BM * RANK];   // scaled a[t,r] for the tile
    __shared__ int sSeg[BM];                          // adapter id per tile row

    const int tid  = threadIdx.x;
    const int row0 = blockIdx.x * BM;
    const int col0 = blockIdx.y * BN;

    // --- LoRA activation tile: async copy global -> LDS (ASYNCcnt path). ---
    // 128x16 floats = 8KB; each of 256 threads moves 32B as 2x b128.
    {
        unsigned    ldsa = (unsigned)(size_t)(void*)(sAc + tid * 8);
        const float* gsa = a_ws + (size_t)row0 * RANK + tid * 8;
        asm volatile("global_load_async_to_lds_b128 %0, %1, off"
                     :: "v"(ldsa), "v"(gsa) : "memory");
        asm volatile("global_load_async_to_lds_b128 %0, %1, off offset:16"
                     :: "v"(ldsa), "v"(gsa) : "memory");
    }
    if (tid < BM) {
        int t = row0 + tid, seg = 0;
        #pragma unroll
        for (int n = 0; n < NADP; ++n) seg += (t >= offsets[n]) ? 1 : 0;
        sSeg[tid] = seg;
    }

    const int lane = tid & 31, wid = tid >> 5;
    const int wm = wid >> 2;          // 0..1 -> 64-row band
    const int wn = wid & 3;           // 0..3 -> 32-col band
    const int lr = lane & 15;
    const int lh = lane >> 4;

    // Staging map: each thread converts+stores half a row (16 bf16) per tile.
    const int srow = tid >> 1;        // 0..127
    const int scol = (tid & 1) * 16;  // 0 or 16

    const float* xbase = x + (size_t)(row0 + srow) * IN_F + scol;
    const float* wbase = W + (size_t)(col0 + srow) * IN_F + scol;
    __bf16* stX0 = &sX[0][srow * LDK + scol];
    __bf16* stX1 = &sX[1][srow * LDK + scol];
    __bf16* stW0 = &sW[0][srow * LDK + scol];
    __bf16* stW1 = &sW[1][srow * LDK + scol];

    v8f acc[4][2];
    #pragma unroll
    for (int f = 0; f < 4; ++f)
        #pragma unroll
        for (int g = 0; g < 2; ++g)
            acc[f][g] = v8f{0.f, 0.f, 0.f, 0.f, 0.f, 0.f, 0.f, 0.f};

    float4 xr0, xr1, xr2, xr3, wr0, wr1, wr2, wr3;

    auto load_g = [&](int k0) {
        const float4* xg = (const float4*)(xbase + k0);
        const float4* wg = (const float4*)(wbase + k0);
        xr0 = xg[0]; xr1 = xg[1]; xr2 = xg[2]; xr3 = xg[3];
        wr0 = wg[0]; wr1 = wg[1]; wr2 = wg[2]; wr3 = wg[3];
    };
    auto store_s = [&](__bf16* dx, __bf16* dw) {
        v8bf px0 = { (__bf16)xr0.x, (__bf16)xr0.y, (__bf16)xr0.z, (__bf16)xr0.w,
                     (__bf16)xr1.x, (__bf16)xr1.y, (__bf16)xr1.z, (__bf16)xr1.w };
        v8bf px1 = { (__bf16)xr2.x, (__bf16)xr2.y, (__bf16)xr2.z, (__bf16)xr2.w,
                     (__bf16)xr3.x, (__bf16)xr3.y, (__bf16)xr3.z, (__bf16)xr3.w };
        v8bf pw0 = { (__bf16)wr0.x, (__bf16)wr0.y, (__bf16)wr0.z, (__bf16)wr0.w,
                     (__bf16)wr1.x, (__bf16)wr1.y, (__bf16)wr1.z, (__bf16)wr1.w };
        v8bf pw1 = { (__bf16)wr2.x, (__bf16)wr2.y, (__bf16)wr2.z, (__bf16)wr2.w,
                     (__bf16)wr3.x, (__bf16)wr3.y, (__bf16)wr3.z, (__bf16)wr3.w };
        *(v8bf*)(dx)     = px0;
        *(v8bf*)(dx + 8) = px1;
        *(v8bf*)(dw)     = pw0;
        *(v8bf*)(dw + 8) = pw1;
    };
    auto compute = [&](const __bf16* bx, const __bf16* bw) {
        // A fragments (16x32 bf16): lane half lh selects the 8-wide K sub-block.
        v16bf af[4];
        #pragma unroll
        for (int f = 0; f < 4; ++f) {
            const __bf16* p = bx + (wm * 64 + f * 16 + lr) * LDK + lh * 8;
            v8bf lo = *(const v8bf*)p;          // K = lh*8 .. +7
            v8bf hi = *(const v8bf*)(p + 16);   // K = 16 + lh*8 .. +7
            af[f] = __builtin_shufflevector(lo, hi, 0,1,2,3,4,5,6,7,8,9,10,11,12,13,14,15);
        }
        // B fragments (32x16 bf16): lane half lh selects the 8-wide K sub-block (same as A).
        v16bf bfr[2];
        #pragma unroll
        for (int g = 0; g < 2; ++g) {
            const __bf16* p = bw + (wn * 32 + g * 16 + lr) * LDK + lh * 8;
            v8bf lo = *(const v8bf*)p;
            v8bf hi = *(const v8bf*)(p + 16);
            bfr[g] = __builtin_shufflevector(lo, hi, 0,1,2,3,4,5,6,7,8,9,10,11,12,13,14,15);
        }
        #pragma unroll
        for (int f = 0; f < 4; ++f)
            #pragma unroll
            for (int g = 0; g < 2; ++g)
                acc[f][g] = __builtin_amdgcn_wmma_f32_16x16x32_bf16(
                    false, af[f], false, bfr[g], (short)0, acc[f][g], false, false);
    };

    // Prologue: fill buffer 0, make async LoRA tile + seg visible.
    load_g(0);
    store_s(stX0, stW0);
    asm volatile("s_wait_asynccnt 0" ::: "memory");
    __syncthreads();

    // Branch-free 2-stage pipeline; statically addressed buffers.
    // Last iteration's second prefetch wraps to k=0 (harmless redundant work).
    for (int k0 = 0; k0 < IN_F; k0 += 2 * BK) {
        load_g(k0 + BK);                    // data for buf1, in flight over compute
        compute(&sX[0][0], &sW[0][0]);      // consume buf0
        store_s(stX1, stW1);
        __syncthreads();

        load_g((k0 + 2 * BK) & (IN_F - 1)); // data for next buf0 (wraps at end)
        compute(&sX[1][0], &sW[1][0]);      // consume buf1
        store_s(stX0, stW0);
        __syncthreads();
    }

    // Epilogue: bias + rank-16 LoRA-B dot, then store fp32.
    // C layout: vgpr i, lanes 0-15 -> row base+i, lanes 16-31 -> row base+8+i.
    const float bv0 = bias[col0 + wn * 32 + lr];
    const float bv1 = bias[col0 + wn * 32 + 16 + lr];
    #pragma unroll
    for (int f = 0; f < 4; ++f) {
        #pragma unroll
        for (int i = 0; i < 8; ++i) {
            const int rl  = wm * 64 + f * 16 + lh * 8 + i;   // row within tile
            const int seg = sSeg[rl];
            const float4* Ap = (const float4*)(sAc + rl * RANK);
            float4 a0 = Ap[0], a1 = Ap[1], a2 = Ap[2], a3 = Ap[3];
            #pragma unroll
            for (int g = 0; g < 2; ++g) {
                const int col = col0 + wn * 32 + g * 16 + lr;
                const float4* Bp = (const float4*)(lB + ((size_t)seg * OUT_F + col) * RANK);
                float4 b0 = Bp[0], b1 = Bp[1], b2 = Bp[2], b3 = Bp[3];
                float s = a0.x*b0.x + a0.y*b0.y + a0.z*b0.z + a0.w*b0.w
                        + a1.x*b1.x + a1.y*b1.y + a1.z*b1.z + a1.w*b1.w
                        + a2.x*b2.x + a2.y*b2.y + a2.z*b2.z + a2.w*b2.w
                        + a3.x*b3.x + a3.y*b3.y + a3.z*b3.z + a3.w*b3.w;
                out[(size_t)(row0 + rl) * OUT_F + col] =
                    acc[f][g][i] + (g ? bv1 : bv0) + s;
            }
        }
    }
}

// ---------------------------------------------------------------------------
extern "C" void kernel_launch(void* const* d_in, const int* in_sizes, int n_in,
                              void* d_out, int out_size, void* d_ws, size_t ws_size,
                              hipStream_t stream) {
    (void)in_sizes; (void)n_in; (void)out_size; (void)ws_size;
    const float* x    = (const float*)d_in[0];
    const float* W    = (const float*)d_in[1];
    const float* bias = (const float*)d_in[2];
    const float* lA   = (const float*)d_in[3];
    const float* lB   = (const float*)d_in[4];
    const int*   offs = (const int*)d_in[5];
    float* out  = (float*)d_out;
    float* a_ws = (float*)d_ws;          // T*R floats = 1 MB scratch

    lora_a_proj<<<T_TOK / 8, 256, 0, stream>>>(x, lA, offs, a_ws);

    dim3 grid(T_TOK / BM, OUT_F / BN);   // 128 x 32 tiles
    lora_gemm<<<grid, 256, 0, stream>>>(x, W, bias, lB, offs, a_ws, out);
}
